// MultiHeadAttention_2662879723916
// MI455X (gfx1250) — matrix operand order repaired and checked by reading, not compiled, not run on hardware
//
#include <hip/hip_runtime.h>
#include <hip/hip_bf16.h>

// MHA for MI455X (gfx1250): bf16 WMMA GEMMs with f32 accumulate.
// cvt -> QKV proj (WMMA + async-LDS B tiles) -> scores (WMMA + async-LDS)
//     -> softmax -> probs@V (WMMA + async-LDS) -> out proj (WMMA)

typedef __attribute__((ext_vector_type(16))) __bf16 v16bf;
typedef __attribute__((ext_vector_type(8)))  float  v8f;

#define DMODEL 1024
#define NH     16
#define DEPTH  64
#define BATCH  4
#define SEQ    2048
#define NROWS  (BATCH * SEQ)   // 8192

// ---------------- low-level helpers ----------------

__device__ __forceinline__ v8f wmma_bf16(v16bf a, v16bf b, v8f c) {
  return __builtin_amdgcn_wmma_f32_16x16x32_bf16(
      /*neg_a=*/false, a, /*neg_b=*/false, b,
      /*c_mod=*/(short)0, c, /*reuse_a=*/false, /*reuse_b=*/false);
}

// Async 16-byte global -> LDS copy (ASYNCcnt tracked).
__device__ __forceinline__ void async_b128(uint32_t lds_off, const void* gaddr) {
  asm volatile("global_load_async_to_lds_b128 %0, %1, off"
               :: "v"(lds_off), "v"(gaddr) : "memory");
}
__device__ __forceinline__ void wait_async_le1() {
  asm volatile("s_wait_asynccnt 0x1" ::: "memory");
}
__device__ __forceinline__ void wait_async_0() {
  asm volatile("s_wait_asynccnt 0x0" ::: "memory");
}

// A fragment (16x32 bf16), tile row-major, leading dim ld.
// Lanes 0-15: row=lane, K 0..7 then 16..23; lanes 16-31: K 8..15 then 24..31.
__device__ __forceinline__ v16bf load_a_bf16(const __bf16* tile, int ld, int lane) {
  const int r = lane & 15, sel = (lane >> 4) & 1;
  const __bf16* p = tile + (size_t)r * ld + sel * 8;
  v16bf a;
  float4* w = reinterpret_cast<float4*>(&a);
  w[0] = *reinterpret_cast<const float4*>(p);
  w[1] = *reinterpret_cast<const float4*>(p + 16);
  return a;
}

// B fragment (32x16 bf16) from a transposed (N,K) row-major buffer (global or LDS).
// Lane group 0: col=lane, K 0..7 then 16..23; group 1: K 8..15 then 24..31.
__device__ __forceinline__ v16bf load_b_bf16(const __bf16* tileT, int ld, int lane) {
  const int c = lane & 15, sel = (lane >> 4) & 1;
  const __bf16* p = tileT + (size_t)c * ld + sel * 8;
  v16bf b;
  float4* w = reinterpret_cast<float4*>(&b);
  w[0] = *reinterpret_cast<const float4*>(p);
  w[1] = *reinterpret_cast<const float4*>(p + 16);
  return b;
}

// A fragment from an f32 row-major tile (probs), converted to bf16.
__device__ __forceinline__ v16bf load_a_f32(const float* tile, int ld, int lane) {
  const int r = lane & 15, sel = (lane >> 4) & 1;
  const float* p = tile + (size_t)r * ld + sel * 8;
  v16bf a;
#pragma unroll
  for (int i = 0; i < 8; ++i) a[i] = (__bf16)p[i];
#pragma unroll
  for (int i = 0; i < 8; ++i) a[8 + i] = (__bf16)p[16 + i];
  return a;
}

// ---------------- conversion kernels ----------------

__global__ __launch_bounds__(256) void transpose_w_bf16_kernel(
    const float* __restrict__ W, __bf16* __restrict__ Wt) {
  const int idx = blockIdx.x * 256 + threadIdx.x;
  const int n = idx >> 10, kk = idx & 1023;
  Wt[idx] = (__bf16)W[kk * DMODEL + n];  // Wt[n,k] = W[k,n]
}

__global__ __launch_bounds__(256) void cvt_bf16_kernel(
    const float* __restrict__ X, __bf16* __restrict__ Y, int n) {
  const int idx = blockIdx.x * 256 + threadIdx.x;
  if (idx < n) Y[idx] = (__bf16)X[idx];
}

// ---------------- projection GEMM: Y = A(8192x1024) @ B(1024x1024) + bias -----------
// B panel (64 cols x 32 k = 4 KB) staged in LDS via async copies, double buffered.
// mode 0: bf16 head-split (b,h,s,d);  mode 1: bf16 (b,h,d,s);  mode 2: f32 row-major.
__global__ __launch_bounds__(256) void gemm_proj_kernel(
    const __bf16* __restrict__ A, const __bf16* __restrict__ Bt,
    const float* __restrict__ bias, __bf16* __restrict__ outB,
    float* __restrict__ outF, int mode) {
  __shared__ __bf16 tileB[2][64 * 32];  // [buf][c_local*32 + kk]

  const int tid = threadIdx.x;
  const int wave = tid >> 5, lane = tid & 31;
  const int row0 = (blockIdx.x * 8 + wave) * 16;
  const int col0 = blockIdx.y * 64;

  // Each of the 256 threads copies one 16B chunk per stage (4 KB total).
  const int crow = tid >> 2;                 // local col 0..63
  const int cchunk = (tid & 3) * 8;          // element offset within 32-k slice
  const __bf16* bsrc = Bt + (size_t)(col0 + crow) * DMODEL + cchunk;
  const uint32_t ldst0 = (uint32_t)(size_t)&tileB[0][crow * 32 + cchunk];
  const uint32_t ldst1 = (uint32_t)(size_t)&tileB[1][crow * 32 + cchunk];

  async_b128(ldst0, bsrc);  // prologue: stage k0 = 0

  v8f acc[4];
#pragma unroll
  for (int t = 0; t < 4; ++t)
#pragma unroll
    for (int i = 0; i < 8; ++i) acc[t][i] = 0.0f;

  const __bf16* arow = A + (size_t)row0 * DMODEL;
  v16bf a_cur = load_a_bf16(arow, DMODEL, lane);

  for (int k0 = 0; k0 < DMODEL; k0 += 32) {
    const int buf = (k0 >> 5) & 1;
    v16bf a_next = a_cur;
    if (k0 + 32 < DMODEL) {
      async_b128(buf ? ldst0 : ldst1, bsrc + k0 + 32);  // stage k0+32
      a_next = load_a_bf16(arow + k0 + 32, DMODEL, lane);
      wait_async_le1();   // current stage complete; next stays in flight
    } else {
      wait_async_0();
    }
    __syncthreads();      // publish current stage to all waves
    const __bf16* bt = &tileB[buf][0];
#pragma unroll
    for (int t = 0; t < 4; ++t) {
      v16bf b = load_b_bf16(bt + t * 16 * 32, 32, lane);
      acc[t] = wmma_bf16(a_cur, b, acc[t]);
    }
    a_cur = a_next;
    __syncthreads();      // all waves done reading before next overwrite
  }

  const int c = lane & 15, sel = (lane >> 4) & 1;
#pragma unroll
  for (int t = 0; t < 4; ++t) {
#pragma unroll
    for (int i = 0; i < 8; ++i) {
      const int row = row0 + sel * 8 + i;
      const int col = col0 + t * 16 + c;
      const float val = acc[t][i] + bias[col];
      if (mode == 0) {
        const int b_ = row >> 11, s = row & 2047, h = col >> 6, d = col & 63;
        outB[((size_t)(b_ * NH + h) * SEQ + s) * DEPTH + d] = (__bf16)val;
      } else if (mode == 1) {
        const int b_ = row >> 11, s = row & 2047, h = col >> 6, d = col & 63;
        outB[((size_t)(b_ * NH + h) * DEPTH + d) * SEQ + s] = (__bf16)val;
      } else {
        outF[(size_t)row * DMODEL + col] = val;
      }
    }
  }
}

// ---------------- scores: S = 0.125 * q @ k^T per (b,h) ----------------
// Full K (=64) B panel (64 cols x 64 k = 8 KB) staged once in LDS.
__global__ __launch_bounds__(256) void attn_scores_kernel(
    const __bf16* __restrict__ qh, const __bf16* __restrict__ kh,
    float* __restrict__ Sout) {
  __shared__ __bf16 tileB[64 * DEPTH];  // [c_local*64 + kk]

  const int tid = threadIdx.x;
  const int wave = tid >> 5, lane = tid & 31;
  const int bh = blockIdx.z;
  const __bf16* q = qh + (size_t)bh * SEQ * DEPTH;
  const __bf16* k = kh + (size_t)bh * SEQ * DEPTH;
  float* S = Sout + (size_t)bh * SEQ * SEQ;
  const int row0 = (blockIdx.y * 8 + wave) * 16;
  const int col0 = blockIdx.x * 64;

  // 8 KB / 256 threads = 32 B per thread (two async b128).
  const int crow = tid >> 2;                 // local col 0..63
  const int cchunk = (tid & 3) * 16;         // element offset 0/16/32/48
  const __bf16* bsrc = k + (size_t)(col0 + crow) * DEPTH + cchunk;
  const uint32_t ldst = (uint32_t)(size_t)&tileB[crow * DEPTH + cchunk];
  async_b128(ldst, bsrc);
  async_b128(ldst + 16, bsrc + 8);
  wait_async_0();
  __syncthreads();

  v8f acc[4];
#pragma unroll
  for (int t = 0; t < 4; ++t)
#pragma unroll
    for (int i = 0; i < 8; ++i) acc[t][i] = 0.0f;

#pragma unroll
  for (int k0 = 0; k0 < DEPTH; k0 += 32) {
    v16bf a = load_a_bf16(q + (size_t)row0 * DEPTH + k0, DEPTH, lane);
#pragma unroll
    for (int t = 0; t < 4; ++t) {
      v16bf b = load_b_bf16(&tileB[t * 16 * DEPTH + k0], DEPTH, lane);
      acc[t] = wmma_bf16(a, b, acc[t]);
    }
  }

  const int c = lane & 15, sel = (lane >> 4) & 1;
#pragma unroll
  for (int t = 0; t < 4; ++t)
#pragma unroll
    for (int i = 0; i < 8; ++i)
      S[(size_t)(row0 + sel * 8 + i) * SEQ + col0 + t * 16 + c] =
          acc[t][i] * 0.125f;  // 1/sqrt(64)
}

// ---------------- softmax over rows of length 2048, in place ----------------
__global__ __launch_bounds__(256) void softmax_kernel(float* __restrict__ W) {
  __shared__ float red[8];
  float* p = W + (size_t)blockIdx.x * SEQ;
  const int t = threadIdx.x;
  float4 x0 = *reinterpret_cast<float4*>(p + t * 8);
  float4 x1 = *reinterpret_cast<float4*>(p + t * 8 + 4);

  float m = fmaxf(fmaxf(fmaxf(x0.x, x0.y), fmaxf(x0.z, x0.w)),
                  fmaxf(fmaxf(x1.x, x1.y), fmaxf(x1.z, x1.w)));
#pragma unroll
  for (int off = 16; off >= 1; off >>= 1) m = fmaxf(m, __shfl_xor(m, off, 32));
  if ((t & 31) == 0) red[t >> 5] = m;
  __syncthreads();
  m = red[0];
#pragma unroll
  for (int i = 1; i < 8; ++i) m = fmaxf(m, red[i]);

  x0.x = __expf(x0.x - m); x0.y = __expf(x0.y - m);
  x0.z = __expf(x0.z - m); x0.w = __expf(x0.w - m);
  x1.x = __expf(x1.x - m); x1.y = __expf(x1.y - m);
  x1.z = __expf(x1.z - m); x1.w = __expf(x1.w - m);

  float s = x0.x + x0.y + x0.z + x0.w + x1.x + x1.y + x1.z + x1.w;
#pragma unroll
  for (int off = 16; off >= 1; off >>= 1) s += __shfl_xor(s, off, 32);
  __syncthreads();
  if ((t & 31) == 0) red[t >> 5] = s;
  __syncthreads();
  s = red[0] + red[1] + red[2] + red[3] + red[4] + red[5] + red[6] + red[7];

  const float inv = 1.0f / s;
  x0.x *= inv; x0.y *= inv; x0.z *= inv; x0.w *= inv;
  x1.x *= inv; x1.y *= inv; x1.z *= inv; x1.w *= inv;
  *reinterpret_cast<float4*>(p + t * 8) = x0;
  *reinterpret_cast<float4*>(p + t * 8 + 4) = x1;
}

// ---------------- attn = probs @ v per (b,h) ----------------
// probs read f32 -> bf16 on the fly; vT tiles (64 x 32 = 4 KB) async-staged in LDS.
__global__ __launch_bounds__(256) void attn_v_kernel(
    const float* __restrict__ Pw, const __bf16* __restrict__ vT,
    __bf16* __restrict__ attnC) {
  __shared__ __bf16 tileB[2][64 * 32];

  const int tid = threadIdx.x;
  const int wave = tid >> 5, lane = tid & 31;
  const int bh = blockIdx.y;
  const int b_ = bh >> 4, h = bh & 15;
  const float* P = Pw + (size_t)bh * SEQ * SEQ;
  const __bf16* vt = vT + (size_t)bh * DEPTH * SEQ;  // (d, s) row-major
  const int row0 = (blockIdx.x * 8 + wave) * 16;

  const int crow = tid >> 2;                 // local col (d) 0..63
  const int cchunk = (tid & 3) * 8;
  const __bf16* bsrc = vt + (size_t)crow * SEQ + cchunk;
  const uint32_t ldst0 = (uint32_t)(size_t)&tileB[0][crow * 32 + cchunk];
  const uint32_t ldst1 = (uint32_t)(size_t)&tileB[1][crow * 32 + cchunk];

  async_b128(ldst0, bsrc);

  v8f acc[4];
#pragma unroll
  for (int t = 0; t < 4; ++t)
#pragma unroll
    for (int i = 0; i < 8; ++i) acc[t][i] = 0.0f;

  const float* prow = P + (size_t)row0 * SEQ;
  v16bf a_cur = load_a_f32(prow, SEQ, lane);

  for (int k0 = 0; k0 < SEQ; k0 += 32) {
    const int buf = (k0 >> 5) & 1;
    v16bf a_next = a_cur;
    if (k0 + 32 < SEQ) {
      async_b128(buf ? ldst0 : ldst1, bsrc + k0 + 32);
      a_next = load_a_f32(prow + k0 + 32, SEQ, lane);
      wait_async_le1();
    } else {
      wait_async_0();
    }
    __syncthreads();
    const __bf16* bt = &tileB[buf][0];
#pragma unroll
    for (int t = 0; t < 4; ++t) {
      v16bf b = load_b_bf16(bt + t * 16 * 32, 32, lane);
      acc[t] = wmma_bf16(a_cur, b, acc[t]);
    }
    a_cur = a_next;
    __syncthreads();
  }

  const int c = lane & 15, sel = (lane >> 4) & 1;
#pragma unroll
  for (int t = 0; t < 4; ++t) {
#pragma unroll
    for (int i = 0; i < 8; ++i) {
      const int s = row0 + sel * 8 + i;
      const int d = t * 16 + c;
      attnC[((size_t)(b_ * SEQ + s)) * DMODEL + h * DEPTH + d] = (__bf16)acc[t][i];
    }
  }
}

// ---------------- host side ----------------

extern "C" void kernel_launch(void* const* d_in, const int* in_sizes, int n_in,
                              void* d_out, int out_size, void* d_ws, size_t ws_size,
                              hipStream_t stream) {
  const float* Q  = (const float*)d_in[0];
  const float* K  = (const float*)d_in[1];
  const float* V  = (const float*)d_in[2];
  const float* Wq = (const float*)d_in[3];
  const float* bq = (const float*)d_in[4];
  const float* Wk = (const float*)d_in[5];
  const float* bk = (const float*)d_in[6];
  const float* Wv = (const float*)d_in[7];
  const float* bv = (const float*)d_in[8];
  const float* Wo = (const float*)d_in[9];
  const float* bo = (const float*)d_in[10];

  float* out     = (float*)d_out;                          // (B,S,DM)
  float* weights = out + (size_t)BATCH * SEQ * DMODEL;     // (B,H,Sq,Sk)

  char* ws = (char*)d_ws;
  const size_t MB = 1024 * 1024;
  __bf16* WqT   = (__bf16*)(ws + 0 * MB);     // (n,k) bf16, 2 MB each
  __bf16* WkT   = (__bf16*)(ws + 2 * MB);
  __bf16* WvT   = (__bf16*)(ws + 4 * MB);
  __bf16* WoT   = (__bf16*)(ws + 6 * MB);
  __bf16* Qb    = (__bf16*)(ws + 8 * MB);     // 16 MB each
  __bf16* Kb    = (__bf16*)(ws + 24 * MB);
  __bf16* Vb    = (__bf16*)(ws + 40 * MB);
  __bf16* qh    = (__bf16*)(ws + 56 * MB);    // (b,h,s,d)
  __bf16* kh    = (__bf16*)(ws + 72 * MB);    // (b,h,s,d)
  __bf16* vT    = (__bf16*)(ws + 88 * MB);    // (b,h,d,s)
  __bf16* attnC = (__bf16*)(ws + 104 * MB);   // (b,s,dm)

  const int NX = NROWS * DMODEL;  // 8388608

  transpose_w_bf16_kernel<<<4096, 256, 0, stream>>>(Wq, WqT);
  transpose_w_bf16_kernel<<<4096, 256, 0, stream>>>(Wk, WkT);
  transpose_w_bf16_kernel<<<4096, 256, 0, stream>>>(Wv, WvT);
  transpose_w_bf16_kernel<<<4096, 256, 0, stream>>>(Wo, WoT);
  cvt_bf16_kernel<<<NX / 256, 256, 0, stream>>>(Q, Qb, NX);
  cvt_bf16_kernel<<<NX / 256, 256, 0, stream>>>(K, Kb, NX);
  cvt_bf16_kernel<<<NX / 256, 256, 0, stream>>>(V, Vb, NX);

  dim3 g1(64, 16);
  gemm_proj_kernel<<<g1, 256, 0, stream>>>(Qb, WqT, bq, qh, nullptr, 0);
  gemm_proj_kernel<<<g1, 256, 0, stream>>>(Kb, WkT, bk, kh, nullptr, 0);
  gemm_proj_kernel<<<g1, 256, 0, stream>>>(Vb, WvT, bv, vT, nullptr, 1);

  dim3 g2(32, 16, 64);
  attn_scores_kernel<<<g2, 256, 0, stream>>>(qh, kh, weights);

  softmax_kernel<<<BATCH * NH * SEQ, 256, 0, stream>>>(weights);

  dim3 g3(16, 64);
  attn_v_kernel<<<g3, 256, 0, stream>>>(weights, vT, attnC);

  gemm_proj_kernel<<<g1, 256, 0, stream>>>(attnC, WoT, bo, nullptr, out, 2);
}
